// Discriminator_29609504539216
// MI455X (gfx1250) — matrix operand order repaired and checked by reading, not compiled, not run on hardware
//
#include <hip/hip_runtime.h>

// ---------------------------------------------------------------------------
// GRU-like recurrent discriminator for MI455X (gfx1250), bf16 WMMA path.
//   B=256, T=96, IN=512, H=1024, KCAT=1536
// Strategy:
//   * convert x, time_delta, Wb, W1, W2, W3 to bf16 once per call (ws scratch)
//   * belta = exp(-relu(td@Wb^T+bb)) via one WMMA GEMM (M=24576,N=1024,K=512)
//   * per time step: kernel A computes u,r (two GEMMs sharing A=[s,x]),
//     kernel B computes n and the fused state update + next-step belta scale
//   * final sigmoid projection kernel
// All matrix math on v_wmma_f32_16x16x32_bf16 with f32 accumulation.
// Step kernels use 16x32 wave tiles (512 waves/launch) to maximize
// latency-hiding occupancy for the serial recurrence; the big belta GEMM
// uses 16x64 wave tiles for better register reuse.
// ---------------------------------------------------------------------------

typedef __attribute__((ext_vector_type(16))) __bf16 v16bf;
typedef __attribute__((ext_vector_type(8)))  float  v8f;

struct alignas(16) U4 { unsigned x, y, z, w; };
union FragU { U4 q[2]; v16bf v; };
static_assert(sizeof(v16bf) == 32, "v16bf must be 32 bytes");

constexpr int Bsz = 256, Tsz = 96, INsz = 512, Hsz = 1024;
constexpr int KCAT = INsz + Hsz;  // 1536

// ----- workspace layout (bytes, all offsets 256B-aligned) -------------------
constexpr size_t SZ_XH    = (size_t)Bsz * Tsz * INsz * 2;  // bf16 x
constexpr size_t SZ_WBH   = (size_t)Hsz * INsz * 2;
constexpr size_t SZ_W     = (size_t)Hsz * KCAT * 2;
constexpr size_t SZ_BELTA = (size_t)Bsz * Tsz * Hsz * 2;

constexpr size_t OFF_XH    = 0;
constexpr size_t OFF_TDH   = OFF_XH + SZ_XH;
constexpr size_t OFF_WBH   = OFF_TDH + SZ_XH;
constexpr size_t OFF_W1H   = OFF_WBH + SZ_WBH;
constexpr size_t OFF_W2H   = OFF_W1H + SZ_W;
constexpr size_t OFF_W3H   = OFF_W2H + SZ_W;
constexpr size_t OFF_BELTA = OFF_W3H + SZ_W;
constexpr size_t OFF_SF    = OFF_BELTA + SZ_BELTA;               // f32 s
constexpr size_t OFF_SH    = OFF_SF + (size_t)Bsz * Hsz * 4;     // bf16 s
constexpr size_t OFF_U     = OFF_SH + (size_t)Bsz * Hsz * 2;     // f32 u
constexpr size_t OFF_RS    = OFF_U + (size_t)Bsz * Hsz * 4;      // bf16 r*s
// total ~109 MB

// ----- helpers --------------------------------------------------------------
__device__ __forceinline__ v16bf loadA16(const __bf16* p) {
  // 16-bit A fragment: elems 0..7 = K[ko..ko+7], elems 8..15 = K[ko+16..ko+23]
  FragU f;
  f.q[0] = *reinterpret_cast<const U4*>(p);
  f.q[1] = *reinterpret_cast<const U4*>(p + 16);
  return f.v;
}
__device__ __forceinline__ v16bf loadB16(const __bf16* p) {
  // 16-bit B fragment: elems 0..7 = K[ko..ko+7], elems 8..15 = K[ko+16..ko+23]
  FragU f;
  f.q[0] = *reinterpret_cast<const U4*>(p);
  f.q[1] = *reinterpret_cast<const U4*>(p + 16);
  return f.v;
}
__device__ __forceinline__ v8f wmma_bf16(v16bf a, v16bf b, v8f c) {
  return __builtin_amdgcn_wmma_f32_16x16x32_bf16(false, a, false, b,
                                                 (short)0, c, false, false);
}
__device__ __forceinline__ float sigmoidf_(float x) {
  return 1.0f / (1.0f + expf(-x));
}

// ----- elementwise utility kernels -----------------------------------------
__global__ void f2bf_kernel(const float* __restrict__ in,
                            __bf16* __restrict__ out, int n) {
  for (int i = blockIdx.x * blockDim.x + threadIdx.x; i < n;
       i += gridDim.x * blockDim.x)
    out[i] = (__bf16)in[i];
}

__global__ void zero_state_kernel(float* __restrict__ sF,
                                  __bf16* __restrict__ sH, int n) {
  for (int i = blockIdx.x * blockDim.x + threadIdx.x; i < n;
       i += gridDim.x * blockDim.x) {
    sF[i] = 0.0f;
    sH[i] = (__bf16)0.0f;
  }
}

// ----- belta GEMM: (B*T, IN) x (IN, H) -> exp(-relu(.)) bf16 ---------------
// wave tile 16x64, block = 4 waves stacked in M (64x64), grid (384,16)
__global__ void __launch_bounds__(128)
belta_kernel(const __bf16* __restrict__ tdH, const __bf16* __restrict__ wbH,
             const float* __restrict__ bb, __bf16* __restrict__ beltaH) {
  const int lane = threadIdx.x & 31;
  const int wave = threadIdx.x >> 5;
  const int l16 = lane & 15, hi = lane >> 4;
  const int mbase = blockIdx.x * 64 + wave * 16;
  const int nbase = blockIdx.y * 64;
  const int rowA = mbase + l16;
  const int koA = hi * 8, koB = hi * 8;

  const v8f vzero = {0.f, 0.f, 0.f, 0.f, 0.f, 0.f, 0.f, 0.f};
  v8f acc[4];
#pragma unroll
  for (int j = 0; j < 4; ++j) acc[j] = vzero;

  const __bf16* arow = tdH + (size_t)rowA * INsz + koA;
#pragma unroll 2
  for (int kk = 0; kk < INsz; kk += 32) {
    v16bf a = loadA16(arow + kk);
#pragma unroll
    for (int j = 0; j < 4; ++j) {
      v16bf b = loadB16(wbH + (size_t)(nbase + j * 16 + l16) * INsz + kk + koB);
      acc[j] = wmma_bf16(a, b, acc[j]);
    }
  }
  const int erow0 = mbase + hi * 8;
#pragma unroll
  for (int j = 0; j < 4; ++j) {
    const int col = nbase + j * 16 + l16;
    const float bc = bb[col];
#pragma unroll
    for (int v = 0; v < 8; ++v) {
      const int idx = (erow0 + v) * Hsz + col;
      beltaH[idx] = (__bf16)expf(-fmaxf(acc[j][v] + bc, 0.0f));
    }
  }
}

// ----- kernel A: u = sig(cat@W1^T+b1), r = sig(cat@W2^T+b2), rs = r*s -------
// wave tile 16x32 (x2 outputs), 4 waves as 2Mx2N -> block 32x64, grid (8,16)
__global__ void __launch_bounds__(128)
step_ur_kernel(const __bf16* __restrict__ sH, const float* __restrict__ sF,
               const __bf16* __restrict__ xH, const __bf16* __restrict__ w1,
               const __bf16* __restrict__ w2, const float* __restrict__ b1,
               const float* __restrict__ b2, float* __restrict__ uF,
               __bf16* __restrict__ rsH, int t) {
  const int lane = threadIdx.x & 31;
  const int wave = threadIdx.x >> 5;
  const int wm = wave & 1, wn = wave >> 1;
  const int l16 = lane & 15, hi = lane >> 4;
  const int mbase = blockIdx.x * 32 + wm * 16;
  const int nbase = blockIdx.y * 64 + wn * 32;
  const int rowA = mbase + l16;
  const int koA = hi * 8, koB = hi * 8;

  const v8f vzero = {0.f, 0.f, 0.f, 0.f, 0.f, 0.f, 0.f, 0.f};
  v8f accU[2], accR[2];
#pragma unroll
  for (int j = 0; j < 2; ++j) { accU[j] = vzero; accR[j] = vzero; }

  const __bf16* aS = sH + (size_t)rowA * Hsz + koA;
  const __bf16* aX = xH + ((size_t)rowA * Tsz + t) * INsz + koA;

  // K phase 1: state part (k = 0..1023)
#pragma unroll 2
  for (int kk = 0; kk < Hsz; kk += 32) {
    v16bf a = loadA16(aS + kk);
#pragma unroll
    for (int j = 0; j < 2; ++j) {
      v16bf b = loadB16(w1 + (size_t)(nbase + j * 16 + l16) * KCAT + kk + koB);
      accU[j] = wmma_bf16(a, b, accU[j]);
    }
#pragma unroll
    for (int j = 0; j < 2; ++j) {
      v16bf b = loadB16(w2 + (size_t)(nbase + j * 16 + l16) * KCAT + kk + koB);
      accR[j] = wmma_bf16(a, b, accR[j]);
    }
  }
  // K phase 2: x part (k = 1024..1535)
#pragma unroll 2
  for (int kk = 0; kk < INsz; kk += 32) {
    v16bf a = loadA16(aX + kk);
#pragma unroll
    for (int j = 0; j < 2; ++j) {
      v16bf b = loadB16(w1 + (size_t)(nbase + j * 16 + l16) * KCAT + Hsz + kk + koB);
      accU[j] = wmma_bf16(a, b, accU[j]);
    }
#pragma unroll
    for (int j = 0; j < 2; ++j) {
      v16bf b = loadB16(w2 + (size_t)(nbase + j * 16 + l16) * KCAT + Hsz + kk + koB);
      accR[j] = wmma_bf16(a, b, accR[j]);
    }
  }
  // epilogue
  const int erow0 = mbase + hi * 8;
#pragma unroll
  for (int j = 0; j < 2; ++j) {
    const int col = nbase + j * 16 + l16;
    const float bb1 = b1[col], bb2 = b2[col];
#pragma unroll
    for (int v = 0; v < 8; ++v) {
      const int idx = (erow0 + v) * Hsz + col;
      const float u = sigmoidf_(accU[j][v] + bb1);
      const float r = sigmoidf_(accR[j][v] + bb2);
      uF[idx] = u;
      rsH[idx] = (__bf16)(r * sF[idx]);
    }
  }
}

// ----- kernel B: n = tanh(cat3@W3^T+b3); state update; scale by next belta --
// wave tile 16x32, 4 waves as 2Mx2N -> block 32x64, grid (8,16)
__global__ void __launch_bounds__(128)
step_n_kernel(const __bf16* __restrict__ rsH, const __bf16* __restrict__ xH,
              const __bf16* __restrict__ w3, const float* __restrict__ b3,
              const float* __restrict__ uF, float* __restrict__ sF,
              __bf16* __restrict__ sH, const __bf16* __restrict__ beltaH,
              int t) {
  const int lane = threadIdx.x & 31;
  const int wave = threadIdx.x >> 5;
  const int wm = wave & 1, wn = wave >> 1;
  const int l16 = lane & 15, hi = lane >> 4;
  const int mbase = blockIdx.x * 32 + wm * 16;
  const int nbase = blockIdx.y * 64 + wn * 32;
  const int rowA = mbase + l16;
  const int koA = hi * 8, koB = hi * 8;

  const v8f vzero = {0.f, 0.f, 0.f, 0.f, 0.f, 0.f, 0.f, 0.f};
  v8f acc[2];
#pragma unroll
  for (int j = 0; j < 2; ++j) acc[j] = vzero;

  const __bf16* aS = rsH + (size_t)rowA * Hsz + koA;
  const __bf16* aX = xH + ((size_t)rowA * Tsz + t) * INsz + koA;

#pragma unroll 2
  for (int kk = 0; kk < Hsz; kk += 32) {
    v16bf a = loadA16(aS + kk);
#pragma unroll
    for (int j = 0; j < 2; ++j) {
      v16bf b = loadB16(w3 + (size_t)(nbase + j * 16 + l16) * KCAT + kk + koB);
      acc[j] = wmma_bf16(a, b, acc[j]);
    }
  }
#pragma unroll 2
  for (int kk = 0; kk < INsz; kk += 32) {
    v16bf a = loadA16(aX + kk);
#pragma unroll
    for (int j = 0; j < 2; ++j) {
      v16bf b = loadB16(w3 + (size_t)(nbase + j * 16 + l16) * KCAT + Hsz + kk + koB);
      acc[j] = wmma_bf16(a, b, acc[j]);
    }
  }
  const int erow0 = mbase + hi * 8;
#pragma unroll
  for (int j = 0; j < 2; ++j) {
    const int col = nbase + j * 16 + l16;
    const float bb3 = b3[col];
#pragma unroll
    for (int v = 0; v < 8; ++v) {
      const int row = erow0 + v;
      const int idx = row * Hsz + col;
      const float nv = tanhf(acc[j][v] + bb3);
      const float u = uF[idx];
      const float s = sF[idx];
      const float st = (1.0f - u) * s + u * nv;  // new raw state
      // fold next step's belta scaling in now (raw state never needed again)
      const float mult =
          (t + 1 < Tsz)
              ? (float)beltaH[((size_t)row * Tsz + (t + 1)) * Hsz + col]
              : 1.0f;
      const float sn = mult * st;
      sF[idx] = sn;
      sH[idx] = (__bf16)sn;
    }
  }
}

// ----- final projection: out = sigmoid(state @ Wo^T + bo) -------------------
__global__ void __launch_bounds__(256)
out_kernel(const float* __restrict__ sF, const float* __restrict__ Wo,
           const float* __restrict__ bo, float* __restrict__ out) {
  const int b = blockIdx.x;
  float sum = 0.0f;
  for (int h = threadIdx.x; h < Hsz; h += blockDim.x)
    sum += sF[b * Hsz + h] * Wo[h];
  __shared__ float red[256];
  red[threadIdx.x] = sum;
  __syncthreads();
  for (int s = blockDim.x >> 1; s > 0; s >>= 1) {
    if ((int)threadIdx.x < s) red[threadIdx.x] += red[threadIdx.x + s];
    __syncthreads();
  }
  if (threadIdx.x == 0) out[b] = sigmoidf_(red[0] + bo[0]);
}

// ---------------------------------------------------------------------------
extern "C" void kernel_launch(void* const* d_in, const int* in_sizes, int n_in,
                              void* d_out, int out_size, void* d_ws,
                              size_t ws_size, hipStream_t stream) {
  (void)in_sizes; (void)n_in; (void)out_size; (void)ws_size;
  const float* x  = (const float*)d_in[0];
  const float* td = (const float*)d_in[1];
  const float* Wb = (const float*)d_in[2];
  const float* bb = (const float*)d_in[3];
  const float* W1 = (const float*)d_in[4];
  const float* b1 = (const float*)d_in[5];
  const float* W2 = (const float*)d_in[6];
  const float* b2 = (const float*)d_in[7];
  const float* W3 = (const float*)d_in[8];
  const float* b3 = (const float*)d_in[9];
  const float* Wo = (const float*)d_in[10];
  const float* bo = (const float*)d_in[11];

  char* ws = (char*)d_ws;
  __bf16* xH     = (__bf16*)(ws + OFF_XH);
  __bf16* tdH    = (__bf16*)(ws + OFF_TDH);
  __bf16* wbH    = (__bf16*)(ws + OFF_WBH);
  __bf16* w1H    = (__bf16*)(ws + OFF_W1H);
  __bf16* w2H    = (__bf16*)(ws + OFF_W2H);
  __bf16* w3H    = (__bf16*)(ws + OFF_W3H);
  __bf16* beltaH = (__bf16*)(ws + OFF_BELTA);
  float*  sF     = (float*)(ws + OFF_SF);
  __bf16* sH     = (__bf16*)(ws + OFF_SH);
  float*  uF     = (float*)(ws + OFF_U);
  __bf16* rsH    = (__bf16*)(ws + OFF_RS);

  auto cvt = [&](const float* src, __bf16* dst, int n) {
    int blocks = (n + 255) / 256;
    if (blocks > 4096) blocks = 4096;
    f2bf_kernel<<<blocks, 256, 0, stream>>>(src, dst, n);
  };
  cvt(x, xH, Bsz * Tsz * INsz);
  cvt(td, tdH, Bsz * Tsz * INsz);
  cvt(Wb, wbH, Hsz * INsz);
  cvt(W1, w1H, Hsz * KCAT);
  cvt(W2, w2H, Hsz * KCAT);
  cvt(W3, w3H, Hsz * KCAT);

  zero_state_kernel<<<256, 256, 0, stream>>>(sF, sH, Bsz * Hsz);

  // belta GEMM: M = B*T = 24576, N = H = 1024
  belta_kernel<<<dim3((Bsz * Tsz) / 64, Hsz / 64), 128, 0, stream>>>(
      tdH, wbH, bb, beltaH);

  // recurrent scan: 512 waves per step-GEMM launch for latency hiding
  const dim3 gstep(Bsz / 32, Hsz / 64);
  for (int t = 0; t < Tsz; ++t) {
    step_ur_kernel<<<gstep, 128, 0, stream>>>(sH, sF, xH, w1H, w2H, b1, b2,
                                              uF, rsH, t);
    step_n_kernel<<<gstep, 128, 0, stream>>>(rsH, xH, w3H, b3, uF, sF, sH,
                                             beltaH, t);
  }

  out_kernel<<<Bsz, 256, 0, stream>>>(sF, Wo, bo, (float*)d_out);
}
